// QuantumBlockchainMemory_74500502716743
// MI455X (gfx1250) — matrix operand order repaired and checked by reading, not compiled, not run on hardware
//
#include <hip/hip_runtime.h>
#include <hip/hip_bf16.h>

// ---------------------------------------------------------------------------
// Problem constants (from reference): B=4, S=2048, D=1024, M=2048, J=8, Q=16
// ---------------------------------------------------------------------------
#define BQ     4
#define SQ     2048
#define DQ     1024
#define MQ     2048
#define BSQ    8192          // B*S rows
#define NHEADS 16
#define HDQ    64

typedef __attribute__((ext_vector_type(16))) __bf16         v16bf;
typedef __attribute__((ext_vector_type(8)))  float          v8f;
typedef __attribute__((ext_vector_type(4)))  unsigned short v4us;
typedef __attribute__((ext_vector_type(8)))  unsigned short v8us;
typedef __attribute__((ext_vector_type(16))) unsigned short v16us;

__device__ __forceinline__ unsigned short f2bf_u16(float f) {
  unsigned u = __builtin_bit_cast(unsigned, f);
  unsigned r = u + 0x7FFFu + ((u >> 16) & 1u);   // round-to-nearest-even
  return (unsigned short)(r >> 16);
}
__device__ __forceinline__ v8f wmma_bf16(v16bf a, v16bf b, v8f c) {
  // D(f32,16x16) = A(bf16,16x32) * B(bf16,32x16) + C
  return __builtin_amdgcn_wmma_f32_16x16x32_bf16(
      /*neg_a=*/false, a, /*neg_b=*/false, b,
      /*c_mod=*/(short)0, c, /*reuse_a=*/false, /*reuse_b=*/false);
}

// Combine two contiguous 8-element halves into one fragment register block.
__device__ __forceinline__ v16bf combine8(v8us lo, v8us hi8) {
  v16us c;
#pragma unroll
  for (int i = 0; i < 8; ++i) { c[i] = lo[i]; c[8 + i] = hi8[i]; }
  return __builtin_bit_cast(v16bf, c);
}
// 32-byte contiguous fragment load (lane-contiguous B fragments).
__device__ __forceinline__ v16bf load16(const unsigned short* __restrict__ p) {
  return __builtin_bit_cast(v16bf, *(const v16us*)p);
}
// A-fragment (16x32 bf16): lane<16 -> row=lane, K={k0..+15};
// lane>=16 -> K={k0+16..+31}. Two 16-byte contiguous loads per lane.
__device__ __forceinline__ v16bf load_a_frag(const unsigned short* __restrict__ Arow,
                                             int k0, int hi) {
  const unsigned short* p = Arow + k0 + hi * 16;
  return combine8(*(const v8us*)p, *(const v8us*)(p + 8));
}

// ---------------------------------------------------------------------------
// Stage 1: quantum addressing + softmax + mem_read (K=8, L2-resident memory),
// emitting mem_read in bf16. One 128-thread block per (b,s) row.
// ---------------------------------------------------------------------------
__global__ void addr_memread_kernel(const float* __restrict__ query,
                                    const float* __restrict__ W_addr,
                                    const float* __restrict__ b_addr,
                                    const float* __restrict__ memory,
                                    unsigned short* __restrict__ mem_read) {
  const int row = blockIdx.x;          // 0..8191
  const int t   = threadIdx.x;         // 0..127
  __shared__ float red[4][128];
  __shared__ float addrs[8];

  const float* qrow = query + (size_t)row * DQ;
  float a0 = 0.f, a1 = 0.f, a2 = 0.f, a3 = 0.f;
  for (int d = t; d < DQ; d += 128) {
    float qv = qrow[d];
    const float* w = W_addr + (size_t)d * 16;   // Q=16 cols, only first 4 used
    a0 += qv * w[0]; a1 += qv * w[1]; a2 += qv * w[2]; a3 += qv * w[3];
  }
  red[0][t] = a0; red[1][t] = a1; red[2][t] = a2; red[3][t] = a3;
  __syncthreads();
  for (int s = 64; s > 0; s >>= 1) {
    if (t < s) {
      red[0][t] += red[0][t + s]; red[1][t] += red[1][t + s];
      red[2][t] += red[2][t + s]; red[3][t] += red[3][t + s];
    }
    __syncthreads();
  }
  if (t == 0) {
    float p[4];
#pragma unroll
    for (int q = 0; q < 4; ++q) {
      float th = red[q][0] + b_addr[q];
      float sn = sinf(0.5f * th);
      p[q] = th > 0.f ? sn * sn : 0.f;
    }
    float sc[8];
    float mx = -1e30f;
#pragma unroll
    for (int j = 0; j < 8; ++j) {
      float s = 0.f;
#pragma unroll
      for (int q = 0; q < 4; ++q) s += ((j >> q) & 1) ? p[q] : (1.f - p[q]);
      sc[j] = s;
      mx = fmaxf(mx, s);
    }
    float sum = 0.f;
#pragma unroll
    for (int j = 0; j < 8; ++j) { sc[j] = __expf(sc[j] - mx); sum += sc[j]; }
    float inv = 1.f / sum;
#pragma unroll
    for (int j = 0; j < 8; ++j) addrs[j] = sc[j] * inv;
  }
  __syncthreads();

  float wj[8];
#pragma unroll
  for (int j = 0; j < 8; ++j) wj[j] = addrs[j];
  unsigned short* orow = mem_read + (size_t)row * MQ;
  // 4 consecutive columns per thread: float4 reads, 8-byte packed bf16 store
  for (int c = t * 4; c < MQ; c += 128 * 4) {
    float s0 = 0.f, s1 = 0.f, s2 = 0.f, s3 = 0.f;
#pragma unroll
    for (int j = 0; j < 8; ++j) {
      float4 m = *(const float4*)(memory + (size_t)j * MQ + c);
      s0 += wj[j] * m.x; s1 += wj[j] * m.y; s2 += wj[j] * m.z; s3 += wj[j] * m.w;
    }
    v4us o;
    o[0] = f2bf_u16(s0); o[1] = f2bf_u16(s1);
    o[2] = f2bf_u16(s2); o[3] = f2bf_u16(s3);
    *(v4us*)(orow + c) = o;
  }
}

// ---------------------------------------------------------------------------
// Weight transpose + fp32 -> bf16: Wt[n*K + k] = bf16(W[k*N + n]).
// Makes every WMMA B-fragment a contiguous 32-byte load.
// ---------------------------------------------------------------------------
__global__ void transpose_cvt_kernel(const float* __restrict__ W,
                                     unsigned short* __restrict__ Wt,
                                     int K, int N) {
  __shared__ unsigned short tile[32][33];
  const int nb = blockIdx.x * 32;
  const int kb = blockIdx.y * 32;
  const int tx = threadIdx.x;          // 0..31
  const int ty = threadIdx.y;          // 0..7
  for (int i = ty; i < 32; i += 8)
    tile[i][tx] = f2bf_u16(W[(size_t)(kb + i) * N + nb + tx]);
  __syncthreads();
  for (int i = ty; i < 32; i += 8)
    Wt[(size_t)(nb + i) * K + kb + tx] = tile[tx][i];
}

// ---------------------------------------------------------------------------
// bf16 GEMM: C(MxN) = A(MxK,row-major) @ Bt(NxK,transposed) + bias.
// Block = 256x64 output tile: 8 waves stacked along M sharing one N-tile
// (identical B addresses -> WGP$ hits). Wave = 32x64 tile: 2 A-frags x
// 4 B-frags -> 8 WMMAs per K-step of 32. B-fragments (single-use) are
// software-pipelined one K-step ahead; A-fragments load at the top of the
// iteration. launch_bounds(...,1) relaxes the VGPR budget -> no spills.
// OMODE: 0 = bf16 row-major, 1 = f32 row-major, 2 = bf16 V-transposed
// ([b][h][d][s] so attention PV fragments are contiguous).
// ---------------------------------------------------------------------------
template <int OMODE>
__global__ __launch_bounds__(256, 1)
void gemm_bf16_wmma_kernel(const unsigned short* __restrict__ A,
                           const unsigned short* __restrict__ Bt,
                           const float* __restrict__ bias,
                           unsigned short* __restrict__ Cb,
                           float* __restrict__ Cf,
                           int M, int N, int K) {
  const int lane = threadIdx.x & 31;
  const int wid  = threadIdx.x >> 5;
  const int nt   = blockIdx.x;                 // N/64 tile (shared by block)
  const int mt   = blockIdx.y * 8 + wid;       // 32-row tile per wave
  if (mt * 32 >= M) return;
  const int l15 = lane & 15;
  const int hi  = lane >> 4;

  const unsigned short* Arow0 = A + (size_t)(mt * 32 + l15) * K;
  const unsigned short* Arow1 = Arow0 + (size_t)16 * K;
  const unsigned short* Bb    = Bt + (size_t)(nt * 64 + l15) * K + hi * 16;
  const size_t bK = (size_t)K;

  // prologue: B fragments for k0 = 0
  v16bf b0 = load16(Bb);
  v16bf b1 = load16(Bb + 16 * bK);
  v16bf b2 = load16(Bb + 32 * bK);
  v16bf b3 = load16(Bb + 48 * bK);

  v8f acc[2][4] = {};
  for (int k0 = 0; k0 < K; k0 += 32) {
    // A fragments for this step (short wait, each reused by 4 WMMAs)
    v16bf a0 = load_a_frag(Arow0, k0, hi);
    v16bf a1 = load_a_frag(Arow1, k0, hi);
    // B fragments for the next step (clamped -> branch-free, always valid)
    const int kn = (k0 + 32 < K) ? k0 + 32 : k0;
    v16bf nb0 = load16(Bb + kn);
    v16bf nb1 = load16(Bb + 16 * bK + kn);
    v16bf nb2 = load16(Bb + 32 * bK + kn);
    v16bf nb3 = load16(Bb + 48 * bK + kn);
    if (k0 + 96 < K) {
      __builtin_prefetch(Arow0 + k0 + 96, 0, 3);   // WGP-scope prefetch
      __builtin_prefetch(Arow1 + k0 + 96, 0, 3);
    }
    // 8 WMMAs; B operands already resident, next-step loads drain underneath
    acc[0][0] = wmma_bf16(a0, b0, acc[0][0]);
    acc[1][0] = wmma_bf16(a1, b0, acc[1][0]);
    acc[0][1] = wmma_bf16(a0, b1, acc[0][1]);
    acc[1][1] = wmma_bf16(a1, b1, acc[1][1]);
    acc[0][2] = wmma_bf16(a0, b2, acc[0][2]);
    acc[1][2] = wmma_bf16(a1, b2, acc[1][2]);
    acc[0][3] = wmma_bf16(a0, b3, acc[0][3]);
    acc[1][3] = wmma_bf16(a1, b3, acc[1][3]);
    b0 = nb0; b1 = nb1; b2 = nb2; b3 = nb3;
  }
  // C layout: lane -> col = lane&15; VGPR r -> row = (lane>>4)*8 + r
#pragma unroll
  for (int half = 0; half < 2; ++half) {
#pragma unroll
    for (int t4 = 0; t4 < 4; ++t4) {
#pragma unroll
      for (int r = 0; r < 8; ++r) {
        int row = mt * 32 + half * 16 + hi * 8 + r;
        int col = nt * 64 + t4 * 16 + l15;
        float v = acc[half][t4][r] + bias[col];
        if (OMODE == 2) {
          // [b][h][d][s]: b=row/S, s=row%S, h=col/64, d=col%64
          int bb = row >> 11, s = row & (SQ - 1);
          int hh = col >> 6,  d = col & (HDQ - 1);
          Cb[(size_t)(((bb * NHEADS + hh) * HDQ + d)) * SQ + s] = f2bf_u16(v);
        } else if (OMODE == 1) {
          Cf[(size_t)row * N + col] = v;
        } else {
          Cb[(size_t)row * N + col] = f2bf_u16(v);
        }
      }
    }
  }
}

// ---------------------------------------------------------------------------
// Flash-style attention, bf16 WMMA, hd=64.
// q/k layout: (B*S, 1024), head h in cols [h*64, h*64+64)  (d-contiguous)
// v layout:   [b][h][d][s]                                  (s-contiguous)
// One wave per (b,h,16-row q-tile); 32-key chunks:
//   4 score WMMAs (K=64) + 4 PV WMMAs (K=32 keys) per chunk.
// Pipelined: V frags for this chunk + K frags for the next chunk are issued
// before the softmax VALU stretch, which hides their latency.
// ---------------------------------------------------------------------------
__global__ __launch_bounds__(128, 1)
void attn_wmma_kernel(const unsigned short* __restrict__ Qm,
                      const unsigned short* __restrict__ Km,
                      const unsigned short* __restrict__ Vt,
                      unsigned short* __restrict__ Om) {
  const int w    = threadIdx.x >> 5;        // wave in block: 0..3
  const int lane = threadIdx.x & 31;
  const int qt   = blockIdx.x * 4 + w;      // q tile: 0..127
  const int h    = blockIdx.y;              // head
  const int b    = blockIdx.z;              // batch
  const int l15  = lane & 15;
  const int hi   = lane >> 4;

  __shared__ unsigned short pt[4][16][40];  // per-wave P tile (16x32, 80B rows)

  const size_t base  = (size_t)(b * SQ) * DQ + (size_t)h * HDQ;
  const size_t vbase = (size_t)((b * NHEADS + h) * HDQ) * SQ;

  // Q fragments: 16 rows x 64 dims -> two K=32 A-fragments, loaded once
  const unsigned short* Qrow = Qm + base + (size_t)(qt * 16 + l15) * DQ;
  const v16bf aq0 = load_a_frag(Qrow, 0,  hi);
  const v16bf aq1 = load_a_frag(Qrow, 32, hi);

  float mrow[8], lrow[8];
#pragma unroll
  for (int r = 0; r < 8; ++r) { mrow[r] = -1e30f; lrow[r] = 0.f; }
  v8f o[4] = {};

  const float scale = 0.125f;               // 1/sqrt(64)

  // K fragments for chunk 0: [tt]{lo,hi} over head dim
  const unsigned short* Kp0 = Km + base + (size_t)l15 * DQ;
  const unsigned short* Kp1 = Km + base + (size_t)(16 + l15) * DQ;
  v16bf bk[4];
  bk[0] = load16(Kp0 + hi * 16);
  bk[1] = load16(Kp0 + 32 + hi * 16);
  bk[2] = load16(Kp1 + hi * 16);
  bk[3] = load16(Kp1 + 32 + hi * 16);

  for (int j0 = 0; j0 < SQ; j0 += 32) {
    // ---- scores: S = Q @ K^T, two 16x16 tiles (keys j0..+15, j0+16..+31)
    v8f sc[2] = {};
    sc[0] = wmma_bf16(aq0, bk[0], sc[0]);
    sc[0] = wmma_bf16(aq1, bk[1], sc[0]);
    sc[1] = wmma_bf16(aq0, bk[2], sc[1]);
    sc[1] = wmma_bf16(aq1, bk[3], sc[1]);

    // ---- issue V frags (this chunk) + K frags (next chunk) before softmax
    const unsigned short* Vp = Vt + vbase + (size_t)j0 + hi * 16;
    v16bf bv[4];
#pragma unroll
    for (int t4 = 0; t4 < 4; ++t4)
      bv[t4] = load16(Vp + (size_t)(t4 * 16 + l15) * SQ);
    const int jn = (j0 + 32 < SQ) ? j0 + 32 : j0;   // clamped, branch-free
    const unsigned short* Kn0 = Km + base + (size_t)(jn + l15) * DQ;
    const unsigned short* Kn1 = Km + base + (size_t)(jn + 16 + l15) * DQ;
    v16bf nbk[4];
    nbk[0] = load16(Kn0 + hi * 16);
    nbk[1] = load16(Kn0 + 32 + hi * 16);
    nbk[2] = load16(Kn1 + hi * 16);
    nbk[3] = load16(Kn1 + 32 + hi * 16);

    // ---- online softmax: rows striped as (hi*8+r) across 16-lane groups
#pragma unroll
    for (int r = 0; r < 8; ++r) {
      float s0 = sc[0][r] * scale;
      float s1 = sc[1][r] * scale;
      float v = fmaxf(s0, s1);
#pragma unroll
      for (int off = 8; off > 0; off >>= 1)
        v = fmaxf(v, __shfl_xor(v, off, 16));
      float mnew  = fmaxf(mrow[r], v);
      float alpha = __expf(mrow[r] - mnew);
      float p0 = __expf(s0 - mnew);
      float p1 = __expf(s1 - mnew);
      float ps = p0 + p1;
#pragma unroll
      for (int off = 8; off > 0; off >>= 1)
        ps += __shfl_xor(ps, off, 16);
      lrow[r] = lrow[r] * alpha + ps;
      mrow[r] = mnew;
#pragma unroll
      for (int t4 = 0; t4 < 4; ++t4) o[t4][r] *= alpha;
      pt[w][hi * 8 + r][l15]      = f2bf_u16(p0);
      pt[w][hi * 8 + r][16 + l15] = f2bf_u16(p1);
    }
    __syncthreads();
    // ---- C-layout P -> A-layout via LDS (two ds_load_b128 per lane)
    const unsigned short* pr = &pt[w][l15][hi * 16];
    v16bf ap = combine8(*(const v8us*)pr, *(const v8us*)(pr + 8));
    // ---- o += P @ V  (fragments already resident)
    o[0] = wmma_bf16(ap, bv[0], o[0]);
    o[1] = wmma_bf16(ap, bv[1], o[1]);
    o[2] = wmma_bf16(ap, bv[2], o[2]);
    o[3] = wmma_bf16(ap, bv[3], o[3]);
    __syncthreads();

    bk[0] = nbk[0]; bk[1] = nbk[1]; bk[2] = nbk[2]; bk[3] = nbk[3];
  }

  unsigned short* Orow = Om + base;
#pragma unroll
  for (int t4 = 0; t4 < 4; ++t4) {
#pragma unroll
    for (int r = 0; r < 8; ++r) {
      float v  = o[t4][r] / lrow[r];
      int row  = qt * 16 + hi * 8 + r;
      int col  = t4 * 16 + l15;
      Orow[(size_t)row * DQ + col] = f2bf_u16(v);
    }
  }
}

// ---------------------------------------------------------------------------
// Host-side orchestration
// ---------------------------------------------------------------------------
extern "C" void kernel_launch(void* const* d_in, const int* in_sizes, int n_in,
                              void* d_out, int out_size, void* d_ws, size_t ws_size,
                              hipStream_t stream) {
  (void)in_sizes; (void)n_in; (void)out_size; (void)ws_size;
  const float* query  = (const float*)d_in[0];
  const float* W_addr = (const float*)d_in[1];
  const float* b_addr = (const float*)d_in[2];
  const float* memory = (const float*)d_in[3];
  const float* W_out  = (const float*)d_in[4];
  const float* b_out  = (const float*)d_in[5];
  const float* Wq     = (const float*)d_in[6];
  const float* bq     = (const float*)d_in[7];
  const float* Wk     = (const float*)d_in[8];
  const float* bk     = (const float*)d_in[9];
  const float* Wv     = (const float*)d_in[10];
  const float* bv     = (const float*)d_in[11];
  const float* Wo     = (const float*)d_in[12];
  const float* bo     = (const float*)d_in[13];
  float* out = (float*)d_out;

  char* ws = (char*)d_ws;
  size_t off = 0;
  auto carve = [&](size_t bytes) -> void* {
    void* p = ws + off;
    off = (off + bytes + 255) & ~(size_t)255;
    return p;
  };
  unsigned short* memread = (unsigned short*)carve((size_t)BSQ * MQ * 2);  // 32 MB
  unsigned short* WoutT   = (unsigned short*)carve((size_t)MQ * DQ * 2);   //  4 MB
  unsigned short* WqT     = (unsigned short*)carve((size_t)DQ * DQ * 2);   //  2 MB
  unsigned short* WkT     = (unsigned short*)carve((size_t)DQ * DQ * 2);
  unsigned short* WvT     = (unsigned short*)carve((size_t)DQ * DQ * 2);
  unsigned short* WoT     = (unsigned short*)carve((size_t)DQ * DQ * 2);
  unsigned short* outb    = (unsigned short*)carve((size_t)BSQ * DQ * 2);  // 16 MB
  unsigned short* qb      = (unsigned short*)carve((size_t)BSQ * DQ * 2);
  unsigned short* kbuf    = (unsigned short*)carve((size_t)BSQ * DQ * 2);
  unsigned short* vt      = (unsigned short*)carve((size_t)BSQ * DQ * 2);
  unsigned short* ctxb    = (unsigned short*)carve((size_t)BSQ * DQ * 2);

  // weight downconversion + transpose (Wt is NxK: B-fragments become
  // contiguous 32-byte loads)
  {
    dim3 tb(32, 8);
    transpose_cvt_kernel<<<dim3(DQ / 32, MQ / 32), tb, 0, stream>>>(W_out, WoutT, MQ, DQ);
    transpose_cvt_kernel<<<dim3(DQ / 32, DQ / 32), tb, 0, stream>>>(Wq, WqT, DQ, DQ);
    transpose_cvt_kernel<<<dim3(DQ / 32, DQ / 32), tb, 0, stream>>>(Wk, WkT, DQ, DQ);
    transpose_cvt_kernel<<<dim3(DQ / 32, DQ / 32), tb, 0, stream>>>(Wv, WvT, DQ, DQ);
    transpose_cvt_kernel<<<dim3(DQ / 32, DQ / 32), tb, 0, stream>>>(Wo, WoT, DQ, DQ);
  }

  // addressing + softmax + mem_read
  addr_memread_kernel<<<BSQ, 128, 0, stream>>>(query, W_addr, b_addr, memory, memread);

  dim3 ggrid(DQ / 64, BSQ / 256);            // block = 256x64 tile, 8 waves

  // out = mem_read @ W_out + b_out   (bf16 out)
  gemm_bf16_wmma_kernel<0><<<ggrid, 256, 0, stream>>>(
      memread, WoutT, b_out, outb, nullptr, BSQ, DQ, MQ);
  // q/k row-major bf16
  gemm_bf16_wmma_kernel<0><<<ggrid, 256, 0, stream>>>(
      outb, WqT, bq, qb, nullptr, BSQ, DQ, DQ);
  gemm_bf16_wmma_kernel<0><<<ggrid, 256, 0, stream>>>(
      outb, WkT, bk, kbuf, nullptr, BSQ, DQ, DQ);
  // v directly into [b][h][d][s] transposed layout
  gemm_bf16_wmma_kernel<2><<<ggrid, 256, 0, stream>>>(
      outb, WvT, bv, vt, nullptr, BSQ, DQ, DQ);

  // attention (flash-style, bf16 WMMA)
  dim3 ag(SQ / 16 / 4, NHEADS, BQ);          // (32, 16, 4), 4 waves/block
  attn_wmma_kernel<<<ag, 128, 0, stream>>>(qb, kbuf, vt, ctxb);

  // final projection straight into d_out (f32)
  gemm_bf16_wmma_kernel<1><<<ggrid, 256, 0, stream>>>(
      ctxb, WoT, bo, nullptr, out, BSQ, DQ, DQ);
}
